// ASquare_81896436400826
// MI455X (gfx1250) — compile-verified, matrix operand order checked by reading, not run on hardware
//
#include <hip/hip_runtime.h>

// ---------------------------------------------------------------------------
// MI455X (gfx1250, wave32) implementation of the "ASquare" attention block.
//   C_IN=256, C_ATT=128, K=2, H=W=256, P = K*H*W = 131072
// GEMMs use v_wmma_f32_16x16x32_bf16 (fp32 accumulate), softmax in fp32.
// Algebraic folding: out = (w_out @ tmpG) @ softmax_g  (saves tmpZ stage).
// Round-2 change: preload A + all 8 B fragments before the 8 WMMAs so the
// scheduler can issue partial s_wait_dscnt and back-to-back v_wmma.
// ---------------------------------------------------------------------------

typedef __attribute__((ext_vector_type(16))) __bf16 v16bf;
typedef __attribute__((ext_vector_type(8)))  float  v8f;

#define PTOT   131072      // K*hw
#define HWD    65536       // hw
#define CATT   128
#define CIN    256

union FragU { v16bf v; uint4 q[2]; };

__device__ __forceinline__ unsigned short bf16rn(float f) {
    unsigned int u = __float_as_uint(f);
    u += 0x7FFFu + ((u >> 16) & 1u);          // round-to-nearest-even
    return (unsigned short)(u >> 16);
}

// Load a 16x32 bf16 WMMA fragment from an LDS matrix stored [row][k],
// row stride `ld` (bf16 elems, multiple of 8 -> rows 16B aligned).
// Per ISA 7.12.2: lane = row0+(lane&15); kbase = (lane<16?0:8);
// elems i=0..7 -> k=kbase+i ; i=8..15 -> k=kbase+16+(i-8)  => two b128 loads.
__device__ __forceinline__ v16bf load_frag(const unsigned short* base,
                                           int row0, int k0, int ld) {
    int lane = threadIdx.x & 31;
    const uint4* p = (const uint4*)(base +
        (size_t)(row0 + (lane & 15)) * ld + (k0 + ((lane >> 4) << 3)));
    FragU f;
    f.q[0] = p[0];   // k = kb .. kb+7
    f.q[1] = p[2];   // k = kb+16 .. kb+23   (+32 bytes)
    return f.v;
}

__device__ __forceinline__ v8f wmma_bf16(v16bf a, v16bf b, v8f c) {
    return __builtin_amdgcn_wmma_f32_16x16x32_bf16(
        false, a, false, b, (short)0, c, false, false);
}

// ---------------------------------------------------------------------------
// K1: feat_{theta,pi,g}(128 x P) = W(128x256) * X(256 x P) + b
//     blockIdx.x = N-tile (128 positions), blockIdx.y = which weight.
// ---------------------------------------------------------------------------
__global__ __launch_bounds__(256) void qkv_kernel(
    const float* __restrict__ content, const float* __restrict__ style,
    const float* __restrict__ w_theta, const float* __restrict__ b_theta,
    const float* __restrict__ w_pi,    const float* __restrict__ b_pi,
    const float* __restrict__ w_g,     const float* __restrict__ b_g,
    float* __restrict__ ws)
{
    __shared__ unsigned short Wl[128][72];   // [m][k] tile, KT=64, pad 8
    __shared__ unsigned short Xl[128][72];   // [n][k] tile

    const int which = blockIdx.y;
    const float* W = (which == 0) ? w_theta : (which == 1) ? w_pi : w_g;
    const float* B = (which == 0) ? b_theta : (which == 1) ? b_pi : b_g;
    float* out = ws + (size_t)which * ((size_t)CATT * PTOT);

    const int pos0 = blockIdx.x * 128;
    const float* src = (pos0 < HWD) ? content : style;
    const int p0 = pos0 & (HWD - 1);

    const int tid  = threadIdx.x;
    const int m0   = (tid >> 5) * 16;          // wave's 16-row strip

    v8f acc[8];
    const v8f vzero = {0.f,0.f,0.f,0.f,0.f,0.f,0.f,0.f};
#pragma unroll
    for (int i = 0; i < 8; ++i) acc[i] = vzero;

    for (int ko = 0; ko < CIN; ko += 64) {
        __syncthreads();
        // weights: Wl[m][c] = bf16(W[m*256 + ko + c])   (coalesced over c)
#pragma unroll
        for (int i = tid; i < 128 * 64; i += 256) {
            int m = i >> 6, c = i & 63;
            Wl[m][c] = bf16rn(W[m * CIN + ko + c]);
        }
        // activations: Xl[n][c] = bf16(x[ko+c][pos0+n]) (coalesced over n)
#pragma unroll
        for (int i = tid; i < 128 * 64; i += 256) {
            int c = i >> 7, n = i & 127;
            Xl[n][c] = bf16rn(src[(size_t)(ko + c) * HWD + p0 + n]);
        }
        __syncthreads();
#pragma unroll
        for (int kk = 0; kk < 64; kk += 32) {
            v16bf a = load_frag(&Wl[0][0], m0, kk, 72);
            v16bf bfr[8];
#pragma unroll
            for (int ns = 0; ns < 8; ++ns)
                bfr[ns] = load_frag(&Xl[0][0], ns * 16, kk, 72);
#pragma unroll
            for (int ns = 0; ns < 8; ++ns)
                acc[ns] = wmma_bf16(a, bfr[ns], acc[ns]);
        }
    }
    // epilogue: bias + store fp32 (row-major [c][P])
    const int lane = tid & 31;
    const int n = lane & 15, mh = (lane < 16) ? 0 : 8;
#pragma unroll
    for (int ns = 0; ns < 8; ++ns)
#pragma unroll
        for (int r = 0; r < 8; ++r) {
            int m = m0 + mh + r;
            out[(size_t)m * PTOT + pos0 + ns * 16 + n] = acc[ns][r] + B[m];
        }
}

// ---------------------------------------------------------------------------
// K2: per-channel max & sum(exp) of feat_pi over all 131072 positions.
// ---------------------------------------------------------------------------
__global__ __launch_bounds__(1024) void pi_reduce(
    const float* __restrict__ featP, float* __restrict__ pmax,
    float* __restrict__ psum)
{
    __shared__ float red[1024];
    const int c = blockIdx.x, tid = threadIdx.x;
    const float* row = featP + (size_t)c * PTOT;

    float m = -3.4e38f;
    for (int i = tid; i < PTOT; i += 1024) m = fmaxf(m, row[i]);
    red[tid] = m; __syncthreads();
    for (int st = 512; st > 0; st >>= 1) {
        if (tid < st) red[tid] = fmaxf(red[tid], red[tid + st]);
        __syncthreads();
    }
    const float M = red[0]; __syncthreads();

    float s = 0.f;
    for (int i = tid; i < PTOT; i += 1024) s += __expf(row[i] - M);
    red[tid] = s; __syncthreads();
    for (int st = 512; st > 0; st >>= 1) {
        if (tid < st) red[tid] += red[tid + st];
        __syncthreads();
    }
    if (tid == 0) { pmax[c] = M; psum[c] = red[0]; }
}

// ---------------------------------------------------------------------------
// K3: alpha_gathering[pos*128 + c] = exp(featP[c,pos]-max[c]) / sum[c]
//     (transposed write via LDS tile: 128 channels x 32 positions)
// ---------------------------------------------------------------------------
__global__ __launch_bounds__(256) void pi_softmax_tr(
    const float* __restrict__ featP, const float* __restrict__ pmax,
    const float* __restrict__ psum, float* __restrict__ outPi)
{
    __shared__ float T[128][33];
    const int pos0 = blockIdx.x * 32;
    const int tid = threadIdx.x;
#pragma unroll
    for (int i = tid; i < 128 * 32; i += 256) {
        int c = i >> 5, px = i & 31;
        T[c][px] = __expf(featP[(size_t)c * PTOT + pos0 + px] - pmax[c]) *
                   (1.0f / psum[c]);
    }
    __syncthreads();
#pragma unroll
    for (int i = tid; i < 128 * 32; i += 256) {
        int px = i >> 7, c = i & 127;
        outPi[(size_t)(pos0 + px) * CATT + c] = T[c][px];
    }
}

// ---------------------------------------------------------------------------
// K4: alpha_distribute[c*P + pos] = softmax over c of featG[:,pos]
// ---------------------------------------------------------------------------
__global__ __launch_bounds__(256) void g_softmax(
    const float* __restrict__ featG, float* __restrict__ outG)
{
    const int pos = blockIdx.x * 256 + threadIdx.x;
    float m = -3.4e38f;
#pragma unroll 4
    for (int c = 0; c < CATT; ++c)
        m = fmaxf(m, featG[(size_t)c * PTOT + pos]);
    float s = 0.f;
#pragma unroll 4
    for (int c = 0; c < CATT; ++c)
        s += __expf(featG[(size_t)c * PTOT + pos] - m);
    const float inv = 1.0f / s;
#pragma unroll 4
    for (int c = 0; c < CATT; ++c)
        outG[(size_t)c * PTOT + pos] =
            __expf(featG[(size_t)c * PTOT + pos] - m) * inv;
}

__global__ void zero_kernel(float* __restrict__ p, int n) {
    int i = blockIdx.x * 256 + threadIdx.x;
    if (i < n) p[i] = 0.f;
}

// ---------------------------------------------------------------------------
// K5: tmpG(128x128) += featT(128 x Kc) * alphaPi(Kc x 128), split-K over
//     256 blocks of Kc=512 positions, fp32 global atomic accumulation.
// ---------------------------------------------------------------------------
__global__ __launch_bounds__(256) void gemm_tmpG(
    const float* __restrict__ featT, const float* __restrict__ alphaPi,
    float* __restrict__ tmpG)
{
    __shared__ unsigned short Al[128][40];   // [m][k] , KT=32, pad 8
    __shared__ unsigned short Bl[128][40];   // [n][k]

    const int k0 = blockIdx.x * 512;
    const int tid = threadIdx.x;
    const int m0 = (tid >> 5) * 16;

    v8f acc[8];
    const v8f vzero = {0.f,0.f,0.f,0.f,0.f,0.f,0.f,0.f};
#pragma unroll
    for (int i = 0; i < 8; ++i) acc[i] = vzero;

    for (int kk = 0; kk < 512; kk += 32) {
        __syncthreads();
#pragma unroll
        for (int i = tid; i < 128 * 32; i += 256) {   // A: featT rows
            int m = i >> 5, j = i & 31;
            Al[m][j] = bf16rn(featT[(size_t)m * PTOT + k0 + kk + j]);
        }
#pragma unroll
        for (int i = tid; i < 128 * 32; i += 256) {   // B: alphaPi (pos-major)
            int j = i >> 7, n = i & 127;
            Bl[n][j] = bf16rn(alphaPi[(size_t)(k0 + kk + j) * CATT + n]);
        }
        __syncthreads();
        v16bf a = load_frag(&Al[0][0], m0, 0, 40);
        v16bf bfr[8];
#pragma unroll
        for (int ns = 0; ns < 8; ++ns)
            bfr[ns] = load_frag(&Bl[0][0], ns * 16, 0, 40);
#pragma unroll
        for (int ns = 0; ns < 8; ++ns)
            acc[ns] = wmma_bf16(a, bfr[ns], acc[ns]);
    }
    const int lane = tid & 31;
    const int n = lane & 15, mh = (lane < 16) ? 0 : 8;
#pragma unroll
    for (int ns = 0; ns < 8; ++ns)
#pragma unroll
        for (int r = 0; r < 8; ++r)
            atomicAdd(&tmpG[(m0 + mh + r) * CATT + ns * 16 + n], acc[ns][r]);
}

// ---------------------------------------------------------------------------
// K6: W2(256x128) = w_out(256x128) @ tmpG(128x128)   (tiny, fp32 FMA)
// ---------------------------------------------------------------------------
__global__ __launch_bounds__(256) void w2_gemm(
    const float* __restrict__ w_out, const float* __restrict__ tmpG,
    float* __restrict__ W2)
{
    const int idx = blockIdx.x * 256 + threadIdx.x;   // 32768 outputs
    const int o = idx >> 7, c = idx & 127;
    float s = 0.f;
#pragma unroll 8
    for (int k = 0; k < CATT; ++k)
        s = fmaf(w_out[o * CATT + k], tmpG[k * CATT + c], s);
    W2[idx] = s;
}

// ---------------------------------------------------------------------------
// K7: out(256 x P) = W2 @ alphaG + b_out ; split into content/style planes.
//     blockIdx.x = N-tile (128 pos), blockIdx.y = output-channel half.
// ---------------------------------------------------------------------------
__global__ __launch_bounds__(256) void final_gemm(
    const float* __restrict__ W2, const float* __restrict__ alphaG,
    const float* __restrict__ b_out, float* __restrict__ out)
{
    __shared__ unsigned short Al[128][72];   // [o][k], KT=64
    __shared__ unsigned short Bl[128][72];   // [n][k]

    const int pos0 = blockIdx.x * 128;
    const int oh = blockIdx.y;                       // channel half
    const float* A = W2 + (size_t)oh * 128 * CATT;
    const int tid = threadIdx.x;
    const int m0 = (tid >> 5) * 16;

    v8f acc[8];
    const v8f vzero = {0.f,0.f,0.f,0.f,0.f,0.f,0.f,0.f};
#pragma unroll
    for (int i = 0; i < 8; ++i) acc[i] = vzero;

    for (int ko = 0; ko < CATT; ko += 64) {
        __syncthreads();
#pragma unroll
        for (int i = tid; i < 128 * 64; i += 256) {
            int r = i >> 6, c = i & 63;
            Al[r][c] = bf16rn(A[r * CATT + ko + c]);
        }
#pragma unroll
        for (int i = tid; i < 128 * 64; i += 256) {
            int c = i >> 7, n = i & 127;
            Bl[n][c] = bf16rn(alphaG[(size_t)(ko + c) * PTOT + pos0 + n]);
        }
        __syncthreads();
#pragma unroll
        for (int kk = 0; kk < 64; kk += 32) {
            v16bf a = load_frag(&Al[0][0], m0, kk, 72);
            v16bf bfr[8];
#pragma unroll
            for (int ns = 0; ns < 8; ++ns)
                bfr[ns] = load_frag(&Bl[0][0], ns * 16, kk, 72);
#pragma unroll
            for (int ns = 0; ns < 8; ++ns)
                acc[ns] = wmma_bf16(a, bfr[ns], acc[ns]);
        }
    }
    // out[kbatch*2^24 + o*65536 + p] ; tile never crosses batch boundary
    const int kb = pos0 >> 16;
    const int pp = pos0 & (HWD - 1);
    const int lane = tid & 31;
    const int n = lane & 15, mh = (lane < 16) ? 0 : 8;
#pragma unroll
    for (int ns = 0; ns < 8; ++ns)
#pragma unroll
        for (int r = 0; r < 8; ++r) {
            int o = oh * 128 + m0 + mh + r;
            out[(size_t)kb * ((size_t)CIN * HWD) + (size_t)o * HWD +
                pp + ns * 16 + n] = acc[ns][r] + b_out[o];
        }
}

// ---------------------------------------------------------------------------
extern "C" void kernel_launch(void* const* d_in, const int* in_sizes, int n_in,
                              void* d_out, int out_size, void* d_ws,
                              size_t ws_size, hipStream_t stream)
{
    const float* content = (const float*)d_in[0];
    const float* style   = (const float*)d_in[1];
    const float* w_theta = (const float*)d_in[2];
    const float* b_theta = (const float*)d_in[3];
    const float* w_pi    = (const float*)d_in[4];
    const float* b_pi    = (const float*)d_in[5];
    const float* w_g     = (const float*)d_in[6];
    const float* b_g     = (const float*)d_in[7];
    const float* w_out   = (const float*)d_in[8];
    const float* b_out   = (const float*)d_in[9];

    float* ws    = (float*)d_ws;
    float* featT = ws;                              // 128*131072
    float* featP = ws + 16777216ull;                // 128*131072
    float* featG = ws + 33554432ull;                // 128*131072
    float* pmax  = ws + 50331648ull;                // 128
    float* psum  = pmax + 128;                      // 128
    float* tmpG  = pmax + 256;                      // 128*128
    float* W2    = tmpG + 16384;                    // 256*128

    float* out   = (float*)d_out;                   // content|style planes
    float* outPi = out + 33554432ull;               // alpha_gathering (P x 128)
    float* outG  = out + 50331648ull;               // alpha_distribute (128 x P)

    qkv_kernel<<<dim3(1024, 3), 256, 0, stream>>>(
        content, style, w_theta, b_theta, w_pi, b_pi, w_g, b_g, ws);
    pi_reduce<<<128, 1024, 0, stream>>>(featP, pmax, psum);
    pi_softmax_tr<<<4096, 256, 0, stream>>>(featP, pmax, psum, outPi);
    g_softmax<<<512, 256, 0, stream>>>(featG, outG);
    zero_kernel<<<64, 256, 0, stream>>>(tmpG, 16384);
    gemm_tmpG<<<256, 256, 0, stream>>>(featT, outPi, tmpG);
    w2_gemm<<<128, 256, 0, stream>>>(w_out, tmpG, W2);
    final_gemm<<<dim3(1024, 2), 256, 0, stream>>>(W2, outG, b_out, out);
}
